// TransformerBlock_695784702229
// MI455X (gfx1250) — matrix operand order repaired and checked by reading, not compiled, not run on hardware
//
#include <hip/hip_runtime.h>
#include <hip/hip_bf16.h>

// ---------------- problem constants ----------------
#define NB   2
#define CC   128
#define HH   192
#define WW   192
#define HWP  (HH*WW)          // 36864 pixels per plane
#define NHd  4                // heads
#define DH   32               // head dim
#define HID  340
#define HIDP 352              // HID padded to multiple of 32
#define C3   (3*CC)           // 384

typedef _Float16 v16h __attribute__((ext_vector_type(16)));
typedef float    v8f  __attribute__((ext_vector_type(8)));
typedef unsigned int u32x4 __attribute__((ext_vector_type(4)));
typedef int      i32x8 __attribute__((ext_vector_type(8)));
typedef int      i32x4 __attribute__((ext_vector_type(4)));

#if defined(__has_builtin)
#if __has_builtin(__builtin_amdgcn_tensor_load_to_lds)
#define HAVE_TDM 1
#else
#define HAVE_TDM 0
#endif
#else
#define HAVE_TDM 0
#endif

__device__ __forceinline__ v8f wmma_f16(v16h a, v16h b, v8f c) {
  // (neg_a, A, neg_b, B, c_mod, C, reuse_a, reuse_b)
  return __builtin_amdgcn_wmma_f32_16x16x32_f16(false, a, false, b, (short)0, c,
                                                false, false);
}

__device__ __forceinline__ float gelu_exact(float x) {
  return 0.5f * x * (1.0f + erff(x * 0.70710678118654752f));
}

#if HAVE_TDM
// TDM 2D tile load: tile_h rows x tile_w f32 elements, row stride in elements.
// D# group0/group1 packing per CDNA5 ISA 8.3/8.4. 6-arg (clang-23) builtin form.
__device__ __forceinline__ void tdm_load_2d(const void* gaddr, unsigned lds_off,
                                            unsigned tile_w, unsigned tile_h,
                                            unsigned row_stride_elems) {
  unsigned long long ga = (unsigned long long)(uintptr_t)gaddr;
  u32x4 g0;
  g0[0] = 1u;                                           // count=1, user desc
  g0[1] = lds_off;                                      // lds_addr (bytes)
  g0[2] = (unsigned)(ga & 0xffffffffu);                 // global_addr[31:0]
  g0[3] = (unsigned)((ga >> 32) & 0x1ffffffu) | (2u << 30); // [56:32] | type=2
  i32x8 g1;
  g1[0] = (int)(2u << 16);                              // data_size = 4B
  g1[1] = (int)((tile_w & 0xffffu) << 16);              // tensor_dim0[15:0]
  g1[2] = (int)(((tile_w >> 16) & 0xffffu) |            // tensor_dim0[31:16]
                ((tile_h & 0xffffu) << 16));            // tensor_dim1[15:0]
  g1[3] = (int)(((tile_h >> 16) & 0xffffu) |            // tensor_dim1[31:16]
                ((tile_w & 0xffffu) << 16));            // tile_dim0
  g1[4] = (int)(tile_h & 0xffffu);                      // tile_dim1 (tile_dim2=0)
  g1[5] = (int)row_stride_elems;                        // tensor_dim0_stride lo32
  g1[6] = 0;                                            // stride hi | dim1_stride lo
  g1[7] = 0;
  i32x4 z4;
  z4[0] = 0; z4[1] = 0; z4[2] = 0; z4[3] = 0;
  i32x8 z8;
  z8[0] = 0; z8[1] = 0; z8[2] = 0; z8[3] = 0;
  z8[4] = 0; z8[5] = 0; z8[6] = 0; z8[7] = 0;
  __builtin_amdgcn_tensor_load_to_lds(g0, g1, z4, z4, z8, 0);
}
#endif

// ---------------- LayerNorm over channels (per pixel) ----------------
__global__ void ln_kernel(const float* __restrict__ x, const float* __restrict__ w,
                          const float* __restrict__ bb, float* __restrict__ y) {
  int lane = threadIdx.x & 31;
  int wave = threadIdx.x >> 5;
  int P = blockIdx.x * 8 + wave;
  int b = P / HWP, p = P % HWP;
  const float* xb = x + (size_t)b * CC * HWP;
  float v[4];
  float sum = 0.f;
#pragma unroll
  for (int q = 0; q < 4; ++q) {
    v[q] = xb[(size_t)(lane + 32 * q) * HWP + p];
    sum += v[q];
  }
#pragma unroll
  for (int o = 16; o > 0; o >>= 1) sum += __shfl_xor(sum, o, 32);
  float mu = sum * (1.0f / CC);
  float var = 0.f;
#pragma unroll
  for (int q = 0; q < 4; ++q) { float d = v[q] - mu; var += d * d; }
#pragma unroll
  for (int o = 16; o > 0; o >>= 1) var += __shfl_xor(var, o, 32);
  var *= (1.0f / CC);
  float rs = rsqrtf(var + 1e-5f);
  float* yb = y + (size_t)b * CC * HWP;
#pragma unroll
  for (int q = 0; q < 4; ++q) {
    int c = lane + 32 * q;
    yb[(size_t)c * HWP + p] = (v[q] - mu) * rs * w[c] + bb[c];
  }
}

// ---------------- 1x1 conv as WMMA GEMM ----------------
// Requirements: Cin % 32 == 0 (pad inputs/weights instead of predicating).
// Partial Cout tile handled by clamping A rows (loads stay in bounds) and
// guarding only the stores.
#define NTI 4
__global__ void conv1x1_wmma(const float* __restrict__ in, const float* __restrict__ wgt,
                             const float* __restrict__ bias, const float* __restrict__ res,
                             float* __restrict__ out, int Cin, int Cout) {
  int lane = threadIdx.x & 31;
  int wave = threadIdx.x >> 5;
  int b    = blockIdx.z;
  int co0  = blockIdx.y * 16;
  int pix0 = (blockIdx.x * 8 + wave) * (16 * NTI);
  const float* inb  = in  + (size_t)b * Cin  * HWP;
  float*       outb = out + (size_t)b * Cout * HWP;
  const float* resb = res ? (res + (size_t)b * Cout * HWP) : nullptr;

  int m    = lane & 15;
  int half = lane >> 4;

  v8f acc0, acc1, acc2, acc3;
#pragma unroll
  for (int e = 0; e < 8; ++e) { acc0[e] = 0.f; acc1[e] = 0.f; acc2[e] = 0.f; acc3[e] = 0.f; }

  // clamp A row so weight loads are unconditional and in-bounds
  int rowA = co0 + m;
  if (rowA >= Cout) rowA = Cout - 1;
  const float* wrow = wgt + (size_t)rowA * Cin;

  int Kc = Cin >> 5;
  for (int kt = 0; kt < Kc; ++kt) {
    int k0 = kt * 32;
    if (kt + 1 < Kc)
      __builtin_prefetch(&inb[(size_t)(k0 + 32) * HWP + pix0], 0, 3);

    // A fragment: weights 16(Cout) x 32(Cin), row-major
    v16h a;
#pragma unroll
    for (int v = 0; v < 8; ++v) {
      int kb = (v < 4 ? 2 * v : 16 + 2 * (v - 4)) + half * 8;
      a[2 * v]     = (_Float16)wrow[k0 + kb];
      a[2 * v + 1] = (_Float16)wrow[k0 + kb + 1];
    }

    // B fragments: activations 32(Cin) x 16(pixels), pixels contiguous
#pragma unroll
    for (int t = 0; t < NTI; ++t) {
      v16h bf;
      int n = pix0 + t * 16 + m;
      const float* ibn = inb + (size_t)(k0 + half * 8) * HWP + n;
#pragma unroll
      for (int v = 0; v < 8; ++v) {
        int kr = (v < 4 ? 2 * v : 16 + 2 * (v - 4));
        bf[2 * v]     = (_Float16)ibn[(size_t)kr * HWP];
        bf[2 * v + 1] = (_Float16)ibn[(size_t)(kr + 1) * HWP];
      }
      if (t == 0)      acc0 = wmma_f16(a, bf, acc0);
      else if (t == 1) acc1 = wmma_f16(a, bf, acc1);
      else if (t == 2) acc2 = wmma_f16(a, bf, acc2);
      else             acc3 = wmma_f16(a, bf, acc3);
    }
  }

  // store D: row = v + 8*half, col = lane&15
#pragma unroll
  for (int v = 0; v < 8; ++v) {
    int row = co0 + v + 8 * half;
    if (row >= Cout) continue;
    float bs = bias[row];
    float* orow = outb + (size_t)row * HWP;
    float f0 = acc0[v], f1 = acc1[v], f2 = acc2[v], f3 = acc3[v];
    if (resb) {
      const float* rrow = resb + (size_t)row * HWP;
      f0 += rrow[pix0 + m];
      f1 += rrow[pix0 + 16 + m];
      f2 += rrow[pix0 + 32 + m];
      f3 += rrow[pix0 + 48 + m];
    }
    orow[pix0 + m]      = f0 + bs;
    orow[pix0 + 16 + m] = f1 + bs;
    orow[pix0 + 32 + m] = f2 + bs;
    orow[pix0 + 48 + m] = f3 + bs;
  }
}

// ---------------- depthwise 3x3, SAME ----------------
__global__ void dwconv3_kernel(const float* __restrict__ in, const float* __restrict__ w,
                               const float* __restrict__ bias, float* __restrict__ out,
                               int total, int Cch) {
  int idx = blockIdx.x * 256 + threadIdx.x;
  if (idx >= total) return;
  int p  = idx % HWP;
  int ch = (idx / HWP) % Cch;
  int i = p / WW, j = p % WW;
  const float* wp   = w + ch * 9;
  const float* base = in + (size_t)(idx - p);
  float acc = bias[ch];
#pragma unroll
  for (int di = -1; di <= 1; ++di)
#pragma unroll
    for (int dj = -1; dj <= 1; ++dj) {
      int ii = i + di, jj = j + dj;
      if (ii >= 0 && ii < HH && jj >= 0 && jj < WW)
        acc += wp[(di + 1) * 3 + (dj + 1)] * base[ii * WW + jj];
    }
  out[idx] = acc;
}

// ---------------- axial attention (one workgroup per (b, head, line)) ------
// LDS: f32 staging for Q,K,V (TDM-loaded) + f16 Q/K/V + f16 S(192x192).
#define SM_QS 0
#define SM_KS (DH*WW*4)            // 24576
#define SM_VS (2*DH*WW*4)          // 49152
#define SM_F16 (3*DH*WW*4)         // 73728
#define SM_S  (SM_F16 + 3*DH*WW*2) // 110592
#define SMEM_ATT (SM_S + WW*WW*2)  // 184320

__global__ void attn_wmma(const float* __restrict__ qkv, const float* __restrict__ factor_p,
                          float* __restrict__ out) {
  extern __shared__ char smc[];
  float*    Qs = (float*)(smc + SM_QS);
  float*    Ks = (float*)(smc + SM_KS);
  float*    Vs = (float*)(smc + SM_VS);
  _Float16* Qh = (_Float16*)(smc + SM_F16);
  _Float16* Kh = Qh + DH * WW;
  _Float16* Vh = Kh + DH * WW;
  _Float16* S  = (_Float16*)(smc + SM_S);

  int wg  = blockIdx.x;
  int row = wg % HH;
  int g   = (wg / HH) % NHd;
  int b   = wg / (HH * NHd);
  float factor = factor_p[0];
  int tid = threadIdx.x;

  size_t baseQ = ((((size_t)b * 3 + 0) * NHd + g) * DH) * HWP + (size_t)row * WW;
  size_t baseK = ((((size_t)b * 3 + 1) * NHd + g) * DH) * HWP + (size_t)row * WW;
  size_t baseV = ((((size_t)b * 3 + 2) * NHd + g) * DH) * HWP + (size_t)row * WW;

  // phase 0a: stage raw f32 Q/K/V tiles (DH x WW, row stride HWP) into LDS
#if HAVE_TDM
  if (tid < 32) {                       // wave 0 drives the Tensor Data Mover
    tdm_load_2d(qkv + baseQ, SM_QS, WW, DH, HWP);
    tdm_load_2d(qkv + baseK, SM_KS, WW, DH, HWP);
    tdm_load_2d(qkv + baseV, SM_VS, WW, DH, HWP);
    __builtin_amdgcn_s_wait_tensorcnt(0);
  }
#else
  for (int idx = tid; idx < DH * WW; idx += 256) {
    int d = idx / WW, i = idx % WW;
    Qs[idx] = qkv[baseQ + (size_t)d * HWP + i];
    Ks[idx] = qkv[baseK + (size_t)d * HWP + i];
    Vs[idx] = qkv[baseV + (size_t)d * HWP + i];
  }
#endif
  __syncthreads();

  // phase 0b: l2-normalize q,k per position; convert to f16
  for (int i = tid; i < WW; i += blockDim.x) {
    float qs = 0.f, ks = 0.f;
    for (int d = 0; d < DH; ++d) {
      float qv = Qs[d * WW + i];
      float kv = Ks[d * WW + i];
      qs += qv * qv; ks += kv * kv;
    }
    float qsc = 1.0f / fmaxf(sqrtf(qs), 1e-12f);
    float ksc = 1.0f / fmaxf(sqrtf(ks), 1e-12f);
    for (int d = 0; d < DH; ++d) {
      Qh[d * WW + i] = (_Float16)(Qs[d * WW + i] * qsc);
      Kh[d * WW + i] = (_Float16)(Ks[d * WW + i] * ksc);
      Vh[d * WW + i] = (_Float16)(Vs[d * WW + i]);
    }
  }
  __syncthreads();

  int wave = tid >> 5, lane = tid & 31;
  int m = lane & 15, half = lane >> 4;

  // phase 1: S = Q^T K  (M=192,N=192,K=32 -> 144 tiles, one WMMA each)
  for (int t = wave; t < 144; t += 8) {
    int i0 = (t / 12) * 16, j0 = (t % 12) * 16;
    v16h a, bf;
#pragma unroll
    for (int v = 0; v < 8; ++v) {
      int kb = (v < 4 ? 2 * v : 16 + 2 * (v - 4)) + half * 8;
      a[2 * v]     = Qh[kb * WW + i0 + m];
      a[2 * v + 1] = Qh[(kb + 1) * WW + i0 + m];
      int k = kb;
      bf[2 * v]     = Kh[k * WW + j0 + m];
      bf[2 * v + 1] = Kh[(k + 1) * WW + j0 + m];
    }
    v8f c;
#pragma unroll
    for (int e = 0; e < 8; ++e) c[e] = 0.f;
    c = wmma_f16(a, bf, c);
#pragma unroll
    for (int v = 0; v < 8; ++v)
      S[(i0 + v + 8 * half) * WW + j0 + m] = (_Float16)(c[v] * factor);
  }
  __syncthreads();

  // phase 2: softmax over key axis (one thread per query row)
  for (int r = tid; r < WW; r += blockDim.x) {
    _Float16* srow = S + r * WW;
    float mx = -1e30f;
    for (int j = 0; j < WW; ++j) mx = fmaxf(mx, (float)srow[j]);
    float sum = 0.f;
    for (int j = 0; j < WW; ++j) {
      float e = __expf((float)srow[j] - mx);
      sum += e;
      srow[j] = (_Float16)e;
    }
    float inv = 1.0f / sum;
    for (int j = 0; j < WW; ++j) srow[j] = (_Float16)((float)srow[j] * inv);
  }
  __syncthreads();

  // phase 3: O^T = V * S^T (M=32, N=192, K=192 -> 24 tiles x 6 K-steps)
  size_t obase = (((size_t)b * CC) + (size_t)g * DH) * HWP + (size_t)row * WW;
  for (int t = wave; t < 24; t += 8) {
    int d0 = (t / 12) * 16, i0 = (t % 12) * 16;
    v8f c;
#pragma unroll
    for (int e = 0; e < 8; ++e) c[e] = 0.f;
#pragma unroll
    for (int kt = 0; kt < 6; ++kt) {
      int j0 = kt * 32;
      v16h a, bf;
#pragma unroll
      for (int v = 0; v < 8; ++v) {
        int kb = (v < 4 ? 2 * v : 16 + 2 * (v - 4)) + half * 8;
        a[2 * v]     = Vh[(d0 + m) * WW + j0 + kb];
        a[2 * v + 1] = Vh[(d0 + m) * WW + j0 + kb + 1];
        int k = kb;
        bf[2 * v]     = S[(i0 + m) * WW + j0 + k];
        bf[2 * v + 1] = S[(i0 + m) * WW + j0 + k + 1];
      }
      c = wmma_f16(a, bf, c);
    }
#pragma unroll
    for (int v = 0; v < 8; ++v) {
      int d = d0 + v + 8 * half;
      out[obase + (size_t)d * HWP + i0 + m] = c[v];
    }
  }
}

// ---------------- tiled H<->W transpose ----------------
__global__ void transpose_hw(const float* __restrict__ in, float* __restrict__ out) {
  __shared__ float tile[16][17];
  int plane = blockIdx.z;
  const float* ib = in + (size_t)plane * HWP;
  float* ob = out + (size_t)plane * HWP;
  int i0 = blockIdx.y * 16, j0 = blockIdx.x * 16;
  int tx = threadIdx.x, ty = threadIdx.y;
  tile[ty][tx] = ib[(i0 + ty) * WW + j0 + tx];
  __syncthreads();
  ob[(j0 + ty) * WW + i0 + tx] = tile[tx][ty];
}

// Y = x + T^T  (fused transpose-back + residual)
__global__ void add_transpose_hw(const float* __restrict__ x, const float* __restrict__ T,
                                 float* __restrict__ Y) {
  __shared__ float tile[16][17];
  int plane = blockIdx.z;
  const float* tb = T + (size_t)plane * HWP;
  const float* xb = x + (size_t)plane * HWP;
  float* yb = Y + (size_t)plane * HWP;
  int i0 = blockIdx.y * 16, j0 = blockIdx.x * 16;
  int tx = threadIdx.x, ty = threadIdx.y;
  tile[ty][tx] = tb[(i0 + ty) * WW + j0 + tx];
  __syncthreads();
  int o = (j0 + ty) * WW + i0 + tx;
  yb[o] = xb[o] + tile[tx][ty];
}

// ---------------- fused GDFN dwconv3 + bias + gelu-gate ----------------
// z: (NB, 2*HID, H, W) -> g: (NB, HIDP, H, W)   (channels >= HID zero-filled
// separately so the following GEMM needs no K guards)
__global__ void dwgate_kernel(const float* __restrict__ z, const float* __restrict__ w,
                              const float* __restrict__ bias, float* __restrict__ g,
                              int total) {
  int idx = blockIdx.x * 256 + threadIdx.x;
  if (idx >= total) return;
  int p = idx % HWP;
  int c = (idx / HWP) % HID;
  int b = idx / (HWP * HID);
  int i = p / WW, j = p % WW;
  const float* z1 = z + ((size_t)b * (2 * HID) + c) * HWP;
  const float* z2 = z + ((size_t)b * (2 * HID) + c + HID) * HWP;
  const float* w1 = w + c * 9;
  const float* w2 = w + (c + HID) * 9;
  float a1 = bias[c], a2 = bias[c + HID];
#pragma unroll
  for (int di = -1; di <= 1; ++di)
#pragma unroll
    for (int dj = -1; dj <= 1; ++dj) {
      int ii = i + di, jj = j + dj;
      if (ii >= 0 && ii < HH && jj >= 0 && jj < WW) {
        int o = ii * WW + jj;
        a1 += w1[(di + 1) * 3 + (dj + 1)] * z1[o];
        a2 += w2[(di + 1) * 3 + (dj + 1)] * z2[o];
      }
    }
  g[((size_t)b * HIDP + c) * HWP + p] = gelu_exact(a2) * a1 + gelu_exact(a1) * a2;
}

// zero the padding channels [HID, HIDP) of the gate buffer
__global__ void zero_gpad_kernel(float* __restrict__ g, int total) {
  int idx = blockIdx.x * 256 + threadIdx.x;
  if (idx >= total) return;
  int p = idx % HWP;
  int c = HID + (idx / HWP) % (HIDP - HID);
  int b = idx / ((HIDP - HID) * HWP);
  g[((size_t)b * HIDP + c) * HWP + p] = 0.f;
}

// zero-pad pout weights: (CC, HID) -> (CC, HIDP)
__global__ void pad_w_kernel(const float* __restrict__ w, float* __restrict__ wp,
                             int total) {
  int idx = blockIdx.x * 256 + threadIdx.x;
  if (idx >= total) return;
  int r = idx / HIDP, k = idx % HIDP;
  wp[idx] = (k < HID) ? w[(size_t)r * HID + k] : 0.f;
}

// ---------------- host orchestration ----------------
extern "C" void kernel_launch(void* const* d_in, const int* in_sizes, int n_in,
                              void* d_out, int out_size, void* d_ws, size_t ws_size,
                              hipStream_t stream) {
  const float* x    = (const float*)d_in[0];
  const float* ln1w = (const float*)d_in[1];
  const float* ln1b = (const float*)d_in[2];
  const float* ln2w = (const float*)d_in[3];
  const float* ln2b = (const float*)d_in[4];
  const float* rq1w = (const float*)d_in[5];
  const float* rq1b = (const float*)d_in[6];
  const float* rq2w = (const float*)d_in[7];
  const float* rq2b = (const float*)d_in[8];
  const float* rq3w = (const float*)d_in[9];
  const float* rq3b = (const float*)d_in[10];
  const float* rfac = (const float*)d_in[11];
  const float* rfw  = (const float*)d_in[12];
  const float* rfb  = (const float*)d_in[13];
  const float* cq1w = (const float*)d_in[14];
  const float* cq1b = (const float*)d_in[15];
  const float* cq2w = (const float*)d_in[16];
  const float* cq2b = (const float*)d_in[17];
  const float* cq3w = (const float*)d_in[18];
  const float* cq3b = (const float*)d_in[19];
  const float* cfac = (const float*)d_in[20];
  const float* cfw  = (const float*)d_in[21];
  const float* cfb  = (const float*)d_in[22];
  const float* pinw = (const float*)d_in[23];
  const float* pinb = (const float*)d_in[24];
  const float* dww  = (const float*)d_in[25];
  const float* dwb  = (const float*)d_in[26];
  const float* poutw= (const float*)d_in[27];
  const float* poutb= (const float*)d_in[28];
  float* out = (float*)d_out;

  // workspace layout (bytes); Z/G alias A/B (disjoint lifetimes)
  const size_t CBYTES  = (size_t)NB * CC * HWP * 4;        //  37.75 MB
  const size_t C3BYTES = 3 * CBYTES;                       // 113.25 MB
  const size_t ZBYTES  = (size_t)NB * 2 * HID * HWP * 4;   // 200.5  MB
  const size_t GPBYTES = (size_t)NB * HIDP * HWP * 4;      // 103.8  MB
  char* ws = (char*)d_ws;
  float* Y  = (float*)(ws);
  float* T  = (float*)(ws + CBYTES);
  float* A  = (float*)(ws + 2 * CBYTES);
  float* Bb = (float*)(ws + 2 * CBYTES + C3BYTES);
  float* Z  = (float*)(ws + 2 * CBYTES);                   // alias A
  float* G  = (float*)(ws + 2 * CBYTES + ZBYTES);
  float* Wp = (float*)(ws + 2 * CBYTES + ZBYTES + GPBYTES);

  dim3 blk(256);

  auto conv = [&](const float* in, const float* w, const float* bs, const float* res,
                  float* o, int Cin, int Cout) {
    dim3 grid(HWP / (16 * NTI * 8), (Cout + 15) / 16, NB);
    conv1x1_wmma<<<grid, blk, 0, stream>>>(in, w, bs, res, o, Cin, Cout);
  };
  auto dwc = [&](const float* in, const float* w, const float* bs, float* o) {
    int total = NB * C3 * HWP;
    dwconv3_kernel<<<dim3((total + 255) / 256), blk, 0, stream>>>(in, w, bs, o, total, C3);
  };

  // ---- attention branch: row attention ----
  ln_kernel<<<dim3(NB * HWP / 8), blk, 0, stream>>>(x, ln1w, ln1b, Y);
  conv(Y, rq1w, rq1b, nullptr, A, CC, C3);
  dwc(A, rq2w, rq2b, Bb);
  dwc(Bb, rq3w, rq3b, A);
  attn_wmma<<<dim3(NB * NHd * HH), blk, SMEM_ATT, stream>>>(A, rfac, T);
  conv(T, rfw, rfb, nullptr, Y, CC, CC);

  // ---- column attention on transposed field ----
  transpose_hw<<<dim3(12, 12, NB * CC), dim3(16, 16), 0, stream>>>(Y, T);
  conv(T, cq1w, cq1b, nullptr, A, CC, C3);
  dwc(A, cq2w, cq2b, Bb);
  dwc(Bb, cq3w, cq3b, A);
  attn_wmma<<<dim3(NB * NHd * HH), blk, SMEM_ATT, stream>>>(A, cfac, Y);
  conv(Y, cfw, cfb, nullptr, T, CC, CC);
  add_transpose_hw<<<dim3(12, 12, NB * CC), dim3(16, 16), 0, stream>>>(x, T, Y);

  // ---- GDFN ----
  ln_kernel<<<dim3(NB * HWP / 8), blk, 0, stream>>>(Y, ln2w, ln2b, T);
  conv(T, pinw, pinb, nullptr, Z, CC, 2 * HID);
  {
    int total = NB * HID * HWP;
    dwgate_kernel<<<dim3((total + 255) / 256), blk, 0, stream>>>(Z, dww, dwb, G, total);
    int ztotal = NB * (HIDP - HID) * HWP;
    zero_gpad_kernel<<<dim3((ztotal + 255) / 256), blk, 0, stream>>>(G, ztotal);
    int wtotal = CC * HIDP;
    pad_w_kernel<<<dim3((wtotal + 255) / 256), blk, 0, stream>>>(poutw, Wp, wtotal);
  }
  conv(G, Wp, poutb, Y /*residual*/, out, HIDP, CC);
}
